// PoolHiddenNet_14242111553568
// MI455X (gfx1250) — matrix operand order repaired and checked by reading, not compiled, not run on hardware
//
#include <hip/hip_runtime.h>

#define NA   384   // agents
#define KDIM 512   // HID1
#define ODIM 1024  // BOT
#define EMBD 64
#define HD   64

typedef __attribute__((ext_vector_type(16))) _Float16 v16h;
typedef __attribute__((ext_vector_type(8)))  _Float16 v8h;
typedef __attribute__((ext_vector_type(4)))  _Float16 v4h;
typedef __attribute__((ext_vector_type(8)))  float    v8f;

union AF { v16h v; v8h h[2]; };

// ---------------------------------------------------------------------------
// Kernel 1: U[b,j] = (e_b + b_se)@W1[0:64] + h_b@W1[64:128]
//           V'[a,j] = e_a@W1[0:64] - b1[j]
// so that layer-1 pre-activation y[a,b,j] = U[b,j] - V'[a,j].
// ---------------------------------------------------------------------------
__global__ __launch_bounds__(KDIM) void prep_uv(
    const float* __restrict__ h_states, const float* __restrict__ end_pos,
    const float* __restrict__ W_se, const float* __restrict__ b_se,
    const float* __restrict__ W1, const float* __restrict__ b1,
    float* __restrict__ U, float* __restrict__ V) {
  __shared__ float se[EMBD], sv[EMBD], sh[HD];
  const int r = blockIdx.x;
  const int j = threadIdx.x;
  if (j < EMBD) {
    float e = end_pos[r * 2 + 0] * W_se[j] + end_pos[r * 2 + 1] * W_se[EMBD + j];
    sv[j] = e;
    se[j] = e + b_se[j];
    sh[j] = h_states[r * HD + j];
  }
  __syncthreads();
  float u = 0.f, v = 0.f;
#pragma unroll 8
  for (int i = 0; i < EMBD; ++i) {
    const float w_e = W1[(size_t)i * KDIM + j];
    u = fmaf(se[i], w_e, u);
    u = fmaf(sh[i], W1[(size_t)(EMBD + i) * KDIM + j], u);
    v = fmaf(sv[i], w_e, v);
  }
  U[(size_t)r * KDIM + j] = u;
  V[(size_t)r * KDIM + j] = v - b1[j];
}

// ---------------------------------------------------------------------------
// Kernel 2: Wt[n][k] = (f16) W2[k][n]   (K-contiguous for B-fragment loads)
// ---------------------------------------------------------------------------
__global__ __launch_bounds__(256) void prep_w2t(const float* __restrict__ W2,
                                                _Float16* __restrict__ Wt) {
  __shared__ _Float16 t[32][33];
  const int n0 = blockIdx.x * 32;
  const int k0 = blockIdx.y * 32;
  for (int i = threadIdx.y; i < 32; i += 8)
    t[threadIdx.x][i] = (_Float16)W2[(size_t)(k0 + i) * ODIM + n0 + threadIdx.x];
  __syncthreads();
  for (int i = threadIdx.y; i < 32; i += 8)
    Wt[(size_t)(n0 + i) * KDIM + k0 + threadIdx.x] = t[i][threadIdx.x];
}

// ---------------------------------------------------------------------------
// Kernel 3: block (a, colT) -> out[a, colT*64 .. +63] =
//   max_b relu( relu(U[b]-V'[a]) @ W2[:, cols] + b2[cols] )
// 4 waves, one 16-col WMMA n-tile each. B (W2 columns) is loop-invariant
// across the 24 b-tiles, so all 16 K-step B fragments live in registers
// (128 VGPRs); the steady-state loop is 2x ds_load_b128 (A) + 1 WMMA.
// ---------------------------------------------------------------------------
#define LPAD 8  // halves of row padding (16B) -> conflict-free ds_load_b128

__global__ __launch_bounds__(128) void pool_gemm(
    const float* __restrict__ U, const float* __restrict__ V,
    const _Float16* __restrict__ Wt, const float* __restrict__ b2,
    float* __restrict__ out) {
  __shared__ _Float16 ylds[16][KDIM + LPAD];  // 16 b-rows x 512 K (f16)
  __shared__ float vlds[KDIM];

  const int a    = blockIdx.x;
  const int colT = blockIdx.y;       // 64 output columns per block
  const int tid  = threadIdx.x;      // 0..127
  const int wave = tid >> 5;         // 0..3
  const int lane = tid & 31;
  const int g    = lane >> 4;        // lane half (VGPR-layout group)
  const int mr   = lane & 15;        // M (A) / N (B,D) index within tile

  // V'[a] -> LDS
  for (int i = tid; i < KDIM; i += 128) vlds[i] = V[(size_t)a * KDIM + i];

  // Preload this wave's B fragments (16 cols x 512 K) into registers.
  // B 32x16 f16 layout: lane (g,mr) holds K = kb+16g..+15 of column mr.
  const int col = colT * 64 + wave * 16 + mr;
  const _Float16* bsrc = Wt + (size_t)col * KDIM + 16 * g;
  v16h bf[16];
#pragma unroll
  for (int ks = 0; ks < 16; ++ks) {
    AF t;
    t.h[0] = *(const v8h*)(bsrc + ks * 32);
    t.h[1] = *(const v8h*)(bsrc + ks * 32 + 8);
    bf[ks] = t.v;
  }
  const float b2v = b2[col];

  float runmax[8];
#pragma unroll
  for (int r = 0; r < 8; ++r) runmax[r] = 0.f;  // max_b relu(z) = max(0, max_b z)

  __syncthreads();  // vlds visible before Y-build reads it

  for (int bt = 0; bt < NA / 16; ++bt) {
    const int b0 = bt * 16;
    // Build Y tile: relu(U[b0+row,k] - V'[a,k]) -> f16 LDS
    for (int idx = tid; idx < 16 * 128; idx += 128) {
      const int row = idx >> 7;
      const int k4  = (idx & 127) << 2;
      const float4 u = *(const float4*)(U + (size_t)(b0 + row) * KDIM + k4);
      v4h yv;
      yv.x = (_Float16)fmaxf(u.x - vlds[k4 + 0], 0.f);
      yv.y = (_Float16)fmaxf(u.y - vlds[k4 + 1], 0.f);
      yv.z = (_Float16)fmaxf(u.z - vlds[k4 + 2], 0.f);
      yv.w = (_Float16)fmaxf(u.w - vlds[k4 + 3], 0.f);
      *(v4h*)(&ylds[row][k4]) = yv;
    }
    __syncthreads();

    // A 16x32 f16 layout: lane (g,mr): halves 0-15 -> K = kb+16g..+15
    // (same K order as B).
    const _Float16* arow = &ylds[mr][0] + 16 * g;
    v8f acc = {};
#pragma unroll
    for (int ks = 0; ks < 16; ++ks) {
      AF af;
      af.h[0] = *(const v8h*)(arow + ks * 32);
      af.h[1] = *(const v8h*)(arow + ks * 32 + 8);
      acc = __builtin_amdgcn_wmma_f32_16x16x32_f16(false, af.v, false, bf[ks],
                                                   (short)0, acc, false, false);
    }
#pragma unroll
    for (int r = 0; r < 8; ++r) runmax[r] = fmaxf(runmax[r], acc[r] + b2v);
    __syncthreads();
  }

  // Reduce over the 16 M-rows: 8 regs (M = r + 8g) then swap lane-halves.
  float m = runmax[0];
#pragma unroll
  for (int r = 1; r < 8; ++r) m = fmaxf(m, runmax[r]);
  const int other = __builtin_amdgcn_ds_swizzle(__builtin_bit_cast(int, m),
                                                0x401F);  // SWAPX16
  m = fmaxf(m, __builtin_bit_cast(float, other));
  if (lane < 16) out[(size_t)a * ODIM + col] = m;
}

// ---------------------------------------------------------------------------
extern "C" void kernel_launch(void* const* d_in, const int* in_sizes, int n_in,
                              void* d_out, int out_size, void* d_ws, size_t ws_size,
                              hipStream_t stream) {
  const float* h_states = (const float*)d_in[0];
  const float* end_pos  = (const float*)d_in[1];
  const float* W_se     = (const float*)d_in[2];
  const float* b_se     = (const float*)d_in[3];
  const float* W1       = (const float*)d_in[4];
  const float* b1       = (const float*)d_in[5];
  const float* W2       = (const float*)d_in[6];
  const float* b2       = (const float*)d_in[7];
  float* out = (float*)d_out;

  char* ws = (char*)d_ws;
  float*    U  = (float*)ws;                                  // 384*512 f32
  float*    V  = U + (size_t)NA * KDIM;                       // 384*512 f32
  _Float16* Wt = (_Float16*)(ws + 2 * (size_t)NA * KDIM * 4); // 1024*512 f16

  prep_uv<<<NA, KDIM, 0, stream>>>(h_states, end_pos, W_se, b_se, W1, b1, U, V);
  prep_w2t<<<dim3(ODIM / 32, KDIM / 32), dim3(32, 8), 0, stream>>>(W2, Wt);
  pool_gemm<<<dim3(NA, ODIM / 64), 128, 0, stream>>>(U, V, Wt, b2, out);
}
